// BiLSTM_CRF_11630771438147
// MI455X (gfx1250) — matrix operand order repaired and checked by reading, not compiled, not run on hardware
//
#include <hip/hip_runtime.h>

#define BB    256
#define TDIM  192
#define BT    (BB * TDIM)      // 49152
#define EDIM  300
#define EPAD  320
#define HDIM  128
#define FH    512
#define NTAGS 13
#define TSTART 0
#define TSTOP  10
#define VOCAB 11626

typedef __attribute__((ext_vector_type(16))) __bf16 v16bf;
typedef __attribute__((ext_vector_type(8)))  float  v8f;

static __device__ __forceinline__ __bf16 f2bf(float f) { return (__bf16)f; }
static __device__ __forceinline__ float sigm(float x) { return 1.0f / (1.0f + expf(-x)); }

// A-fragment K index for 16-bit 16x32 A (contiguous order: k = 16*half + e, matches B fills)
static __device__ __forceinline__ int a_kidx(int e, int half) {
  return 16 * half + e;
}

// low 32 bits of a flat LDS pointer == wave-relative LDS byte offset (aperture mapping)
static __device__ __forceinline__ unsigned lds_off(const void* p) {
  return (unsigned)(unsigned long long)p;
}

// gfx1250 async global->LDS copies (ASYNCcnt-tracked), GVS addressing mode
static __device__ __forceinline__ void async_b32(unsigned lds, unsigned voff, unsigned long long base) {
  asm volatile("global_load_async_to_lds_b32 %0, %1, %2 offset:0"
               :: "v"(lds), "v"(voff), "s"(base) : "memory");
}
static __device__ __forceinline__ void async_b128(unsigned lds, unsigned voff, unsigned long long base) {
  asm volatile("global_load_async_to_lds_b128 %0, %1, %2 offset:0"
               :: "v"(lds), "v"(voff), "s"(base) : "memory");
}
#define WAIT_ASYNC(n) asm volatile("s_wait_asynccnt " #n ::: "memory")

// ---------------- weight / embedding prep: f32 -> bf16 (with padding) ----------------
__global__ __launch_bounds__(256) void prep_pad_bf16(const float* __restrict__ W, __bf16* __restrict__ Wb,
                                                     int rows, int kin, int kpad) {
  int i = blockIdx.x * 256 + threadIdx.x;
  if (i >= rows * kpad) return;
  int n = i / kpad, k = i % kpad;
  Wb[i] = f2bf(k < kin ? W[n * kin + k] : 0.0f);
}
__global__ __launch_bounds__(256) void prep_fc(const float* __restrict__ W, __bf16* __restrict__ Wb) {
  int i = blockIdx.x * 256 + threadIdx.x;          // over 16*256
  if (i >= 16 * 256) return;
  int n = i / 256, k = i % 256;
  Wb[i] = f2bf(n < NTAGS ? W[n * 256 + k] : 0.0f);
}

// ---------------- gx = gather(emb_bf) @ W_ih^T + b : M=BT, N=512, K=320 ----------------
// grid (BT/16, 4), block 256 (8 waves, each wave one 16x16 N-tile of a 16x128 slab).
// A tiles double-buffered in LDS via async copies; B fragments loop-invariant in VGPRs.
__global__ __launch_bounds__(256) void gx_gemm(const int* __restrict__ sentence,
                                               const __bf16* __restrict__ emb_bf,
                                               const __bf16* __restrict__ Wih,
                                               const float* __restrict__ bias,
                                               float* __restrict__ gx) {
  __shared__ __bf16 a_lds[2][16][32];
  __shared__ int words[16];
  const int tid = threadIdx.x, wave = tid >> 5, lane = tid & 31;
  const int half = lane >> 4, cidx = lane & 15;
  const int m0 = blockIdx.x * 16;
  const int n = blockIdx.y * 128 + wave * 16 + cidx;
  const unsigned long long embB = (unsigned long long)emb_bf;

  if (tid < 16) words[tid] = sentence[m0 + tid];

  // preload all B fragments (10 K-chunks, loop-invariant)
  v16bf bfrag[10];
#pragma unroll
  for (int kc = 0; kc < 10; kc++) {
#pragma unroll
    for (int e = 0; e < 16; e++) bfrag[kc][e] = Wih[n * EPAD + kc * 32 + 16 * half + e];
  }
  __syncthreads();

  // per-thread staging slot: one b32 (2 bf16) of the 16x32 tile
  const int sm = tid >> 4, sp = tid & 15;
  const unsigned vbase = (unsigned)(words[sm] * (EPAD * 2) + sp * 4);
  unsigned dst0 = lds_off(&a_lds[0][sm][sp * 2]);
  unsigned dst1 = lds_off(&a_lds[1][sm][sp * 2]);

  async_b32(dst0, vbase, embB);                         // chunk 0 -> buf 0

  v8f acc;
#pragma unroll
  for (int v = 0; v < 8; v++) acc[v] = 0.0f;

  for (int kc = 0; kc < 10; kc++) {
    if (kc + 1 < 10) {                                  // prefetch next chunk
      async_b32((kc & 1) ? dst0 : dst1, vbase + (unsigned)((kc + 1) * 64), embB);
      WAIT_ASYNC(1);
    } else {
      WAIT_ASYNC(0);
    }
    __syncthreads();                                    // tile kc visible everywhere
    v16bf a;
#pragma unroll
    for (int e = 0; e < 16; e++) a[e] = a_lds[kc & 1][cidx][a_kidx(e, half)];
    acc = __builtin_amdgcn_wmma_f32_16x16x32_bf16(false, a, false, bfrag[kc], (short)0, acc, false, false);
    __syncthreads();                                    // reads done before buf reuse
  }
  float bn = bias[n];
#pragma unroll
  for (int v = 0; v < 8; v++) gx[(m0 + v + 8 * half) * FH + n] = acc[v] + bn;
}

// ---------------- LSTM scan: 16 blocks x 16 batch rows; t-sequential ----------------
// Per step: D = h_bf16 x W_hh^T + gx_slab(C).  gx slab for step s+1 async-prefetched
// into the alternate LDS buffer while step s computes.  h kept as bf16 in LDS so A
// fragments are two contiguous ds_load_b128 per fragment.
__global__ __launch_bounds__(256) void lstm_scan(const float* __restrict__ gx,
                                                 const __bf16* __restrict__ Whh,
                                                 const int* __restrict__ seq_len,
                                                 float* __restrict__ h_out,
                                                 int backward) {
  __shared__ __bf16 h_bf[16][HDIM];                     // 4 KB
  __shared__ float  c_lds[16][HDIM];                    // 8 KB
  __shared__ float  g_lds[16][FH];                      // 32 KB
  __shared__ float  gx_lds[2][16][FH];                  // 64 KB (double buffer)
  __shared__ int    len_lds[16];

  const int tid = threadIdx.x, wave = tid >> 5, lane = tid & 31;
  const int half = lane >> 4, cidx = lane & 15;
  const int blk = blockIdx.x;
  const unsigned long long gxB = (unsigned long long)gx;

  for (int i = tid; i < 16 * HDIM; i += 256) { (&h_bf[0][0])[i] = f2bf(0.0f); (&c_lds[0][0])[i] = 0.0f; }
  if (tid < 16) {
    int L = seq_len[blk * 16 + tid];
    len_lds[tid] = L < 1 ? 1 : (L > TDIM ? TDIM : L);
  }

  // loop-invariant B fragments: 4 N-tiles/wave x 4 K-chunks (K=128)
  v16bf bfrag[4][4];
#pragma unroll
  for (int j = 0; j < 4; j++) {
    int nn = (wave * 4 + j) * 16 + cidx;
#pragma unroll
    for (int kc = 0; kc < 4; kc++) {
#pragma unroll
      for (int e = 0; e < 16; e++) bfrag[j][kc][e] = Whh[nn * HDIM + kc * 32 + 16 * half + e];
    }
  }
  __syncthreads();                                      // len_lds / h / c ready

  // per-thread slab staging: 8 x b128 per slab (16 rows x 512 f32 = 32 KB)
  // unit u (0..2047): row mr = u>>7, 16B-column = u&127
  auto issue_slab = [&](int s, int buf) {
#pragma unroll
    for (int j = 0; j < 8; j++) {
      int u = tid + j * 256;
      int mr = u >> 7, col = u & 127;
      int len = len_lds[mr];
      int tt = backward ? (s < len ? len - 1 - s : s) : s;
      unsigned voff = (unsigned)((((blk * 16 + mr) * TDIM + tt) * FH) * 4 + col * 16);
      async_b128(lds_off(&gx_lds[buf][mr][col * 4]), voff, gxB);
    }
  };

  issue_slab(0, 0);

  for (int s = 0; s < TDIM; s++) {
    if (s + 1 < TDIM) {                                 // prefetch slab s+1
      issue_slab(s + 1, (s + 1) & 1);
      WAIT_ASYNC(8);
    } else {
      WAIT_ASYNC(0);
    }
    __syncthreads();                                    // slab s + prev gates visible

    v8f acc[4];
#pragma unroll
    for (int j = 0; j < 4; j++) {                       // C := gx slab tile
      int nn = (wave * 4 + j) * 16 + cidx;
      v8f cfr;
#pragma unroll
      for (int v = 0; v < 8; v++) cfr[v] = gx_lds[s & 1][v + 8 * half][nn];
      acc[j] = cfr;
    }
#pragma unroll
    for (int kc = 0; kc < 4; kc++) {
      v16bf a;
#pragma unroll
      for (int e = 0; e < 16; e++) a[e] = h_bf[cidx][kc * 32 + a_kidx(e, half)];
#pragma unroll
      for (int j = 0; j < 4; j++)
        acc[j] = __builtin_amdgcn_wmma_f32_16x16x32_bf16(false, a, false, bfrag[j][kc],
                                                         (short)0, acc[j], false, false);
    }
#pragma unroll
    for (int j = 0; j < 4; j++) {                       // scatter gate pre-activations
      int nn = (wave * 4 + j) * 16 + cidx;
#pragma unroll
      for (int v = 0; v < 8; v++) g_lds[v + 8 * half][nn] = acc[j][v];
    }
    __syncthreads();                                    // g complete; slab reads done

    for (int i = tid; i < 16 * HDIM; i += 256) {        // gates + state update + store
      int mr = i >> 7, jj = i & (HDIM - 1);
      int len = len_lds[mr];
      int tt = backward ? (s < len ? len - 1 - s : s) : s;
      float outv = 0.0f;
      if (s < len) {
        float gi = g_lds[mr][jj], gf = g_lds[mr][HDIM + jj];
        float gg = g_lds[mr][2 * HDIM + jj], go = g_lds[mr][3 * HDIM + jj];
        float cn = sigm(gf) * c_lds[mr][jj] + sigm(gi) * tanhf(gg);
        float hn = sigm(go) * tanhf(cn);
        c_lds[mr][jj] = cn;
        h_bf[mr][jj] = f2bf(hn);
        outv = hn;
      }
      h_out[((blk * 16 + mr) * TDIM + tt) * HDIM + jj] = outv;
    }
    // next iteration's WAIT + barrier orders these writes before the next WMMA
  }
}

// ---------------- feats = [h_f|h_b] @ W_fc^T + b_fc : N padded 13->16 ----------------
__global__ __launch_bounds__(256) void fc_gemm(const float* __restrict__ hf,
                                               const float* __restrict__ hb,
                                               const __bf16* __restrict__ Wfc,
                                               const float* __restrict__ bfc,
                                               float* __restrict__ feats) {
  const int tid = threadIdx.x, wave = tid >> 5, lane = tid & 31;
  const int half = lane >> 4, cidx = lane & 15;
  const int m0 = (blockIdx.x * 8 + wave) * 16;
  const int mrow = m0 + cidx;

  v16bf bfrag[8];
#pragma unroll
  for (int kc = 0; kc < 8; kc++) {
#pragma unroll
    for (int e = 0; e < 16; e++) bfrag[kc][e] = Wfc[cidx * 256 + kc * 32 + 16 * half + e];
  }

  float bn = (cidx < NTAGS) ? bfc[cidx] : 0.0f;
  v8f acc;
#pragma unroll
  for (int v = 0; v < 8; v++) acc[v] = bn;

#pragma unroll
  for (int kc = 0; kc < 8; kc++) {
    v16bf a;
#pragma unroll
    for (int e = 0; e < 16; e++) {
      int k = kc * 32 + a_kidx(e, half);
      float av = (k < HDIM) ? hf[mrow * HDIM + k] : hb[mrow * HDIM + (k - HDIM)];
      a[e] = f2bf(av);
    }
    acc = __builtin_amdgcn_wmma_f32_16x16x32_bf16(false, a, false, bfrag[kc], (short)0, acc, false, false);
  }
#pragma unroll
  for (int v = 0; v < 8; v++) feats[(m0 + v + 8 * half) * 16 + cidx] = acc[v];
}

// ---------------- CRF forward + gold score: one wave per batch ----------------
__global__ __launch_bounds__(32) void crf_nll(const float* __restrict__ feats,
                                              const int* __restrict__ tags,
                                              const float* __restrict__ trans,
                                              float* __restrict__ nll) {
  __shared__ float tr[NTAGS * NTAGS];
  __shared__ float al[16];
  const int b = blockIdx.x, lane = threadIdx.x;
  for (int i = lane; i < NTAGS * NTAGS; i += 32) tr[i] = trans[i];
  float alpha = (lane == TSTART) ? 0.0f : -10000.0f;
  __syncthreads();

  for (int t = 0; t < TDIM; t++) {
    if (lane < NTAGS) al[lane] = alpha;
    __syncthreads();
    float nxt = 0.0f;
    if (lane < NTAGS) {
      float mx = -1e30f;
      for (int i = 0; i < NTAGS; i++) { float v = al[i] + tr[lane * NTAGS + i]; mx = v > mx ? v : mx; }
      float s = 0.0f;
      for (int i = 0; i < NTAGS; i++) s += expf(al[i] + tr[lane * NTAGS + i] - mx);
      nxt = mx + logf(s) + feats[(b * TDIM + t) * 16 + lane];
    }
    __syncthreads();
    if (lane < NTAGS) alpha = nxt;
  }

  if (lane < NTAGS) al[lane] = alpha + tr[TSTOP * NTAGS + lane];
  __syncthreads();
  if (lane == 0) {
    float mx = -1e30f;
    for (int i = 0; i < NTAGS; i++) mx = al[i] > mx ? al[i] : mx;
    float s = 0.0f;
    for (int i = 0; i < NTAGS; i++) s += expf(al[i] - mx);
    float fwd = mx + logf(s);

    int prev = TSTART;
    float gold = 0.0f;
    for (int t = 0; t < TDIM; t++) {
      int tg = tags[b * TDIM + t];
      gold += tr[tg * NTAGS + prev] + feats[(b * TDIM + t) * 16 + tg];
      prev = tg;
    }
    gold += tr[TSTOP * NTAGS + prev];
    nll[b] = fwd - gold;
  }
}

__global__ __launch_bounds__(256) void reduce_mean(const float* __restrict__ nll, float* __restrict__ out) {
  __shared__ float s[256];
  int tid = threadIdx.x;
  s[tid] = nll[tid];
  __syncthreads();
  for (int off = 128; off > 0; off >>= 1) {
    if (tid < off) s[tid] += s[tid + off];
    __syncthreads();
  }
  if (tid == 0) out[0] = s[0] / 256.0f;
}

extern "C" void kernel_launch(void* const* d_in, const int* in_sizes, int n_in,
                              void* d_out, int out_size, void* d_ws, size_t ws_size,
                              hipStream_t stream) {
  const int* sentence  = (const int*)d_in[0];
  const int* seq_len   = (const int*)d_in[1];
  const int* tags      = (const int*)d_in[2];
  const float* embedding = (const float*)d_in[3];
  const float* W_ih_f  = (const float*)d_in[4];
  const float* W_hh_f  = (const float*)d_in[5];
  const float* b_f     = (const float*)d_in[6];
  const float* W_ih_b  = (const float*)d_in[7];
  const float* W_hh_b  = (const float*)d_in[8];
  const float* b_b     = (const float*)d_in[9];
  const float* W_fc    = (const float*)d_in[10];
  const float* b_fc    = (const float*)d_in[11];
  const float* trans   = (const float*)d_in[12];
  float* out = (float*)d_out;

  char* ws = (char*)d_ws;
  size_t off = 0;
  auto alloc = [&](size_t bytes) -> void* {
    void* p = ws + off;
    off += (bytes + 255) & ~(size_t)255;
    return p;
  };
  float*  gx     = (float*)alloc((size_t)BT * FH * 4);        // ~100.7 MB, reused per dir
  float*  h_f    = (float*)alloc((size_t)BT * HDIM * 4);      // 25.2 MB
  float*  h_b    = (float*)alloc((size_t)BT * HDIM * 4);      // 25.2 MB
  float*  feats  = (float*)alloc((size_t)BT * 16 * 4);        // 3.1 MB
  float*  nll    = (float*)alloc(256 * 4);
  __bf16* emb_bf = (__bf16*)alloc((size_t)VOCAB * EPAD * 2);  // 7.4 MB, K-padded
  __bf16* Wih_bf = (__bf16*)alloc((size_t)FH * EPAD * 2);     // reused per dir
  __bf16* Whh_bf = (__bf16*)alloc((size_t)FH * HDIM * 2);     // reused per dir
  __bf16* Wfc_bf = (__bf16*)alloc((size_t)16 * 256 * 2);

  dim3 gxGrid(BT / 16, FH / 128);

  prep_pad_bf16<<<((VOCAB * EPAD) + 255) / 256, 256, 0, stream>>>(embedding, emb_bf, VOCAB, EDIM, EPAD);
  prep_fc<<<(16 * 256 + 255) / 256, 256, 0, stream>>>(W_fc, Wfc_bf);

  // forward direction
  prep_pad_bf16<<<((FH * EPAD) + 255) / 256, 256, 0, stream>>>(W_ih_f, Wih_bf, FH, EDIM, EPAD);
  prep_pad_bf16<<<((FH * HDIM) + 255) / 256, 256, 0, stream>>>(W_hh_f, Whh_bf, FH, HDIM, HDIM);
  gx_gemm<<<gxGrid, 256, 0, stream>>>(sentence, emb_bf, Wih_bf, b_f, gx);
  lstm_scan<<<BB / 16, 256, 0, stream>>>(gx, Whh_bf, seq_len, h_f, 0);

  // backward direction
  prep_pad_bf16<<<((FH * EPAD) + 255) / 256, 256, 0, stream>>>(W_ih_b, Wih_bf, FH, EDIM, EPAD);
  prep_pad_bf16<<<((FH * HDIM) + 255) / 256, 256, 0, stream>>>(W_hh_b, Whh_bf, FH, HDIM, HDIM);
  gx_gemm<<<gxGrid, 256, 0, stream>>>(sentence, emb_bf, Wih_bf, b_b, gx);
  lstm_scan<<<BB / 16, 256, 0, stream>>>(gx, Whh_bf, seq_len, h_b, 1);

  fc_gemm<<<BT / 128, 256, 0, stream>>>(h_f, h_b, Wfc_bf, b_fc, feats);
  crf_nll<<<BB, 32, 0, stream>>>(feats, tags, trans, nll);
  reduce_mean<<<1, 256, 0, stream>>>(nll, out);
}
